// Linear8bitLt_66005057405785
// MI455X (gfx1250) — compile-verified, matrix operand order checked by reading, not run on hardware
//
#include <hip/hip_runtime.h>
#include <hip/hip_bf16.h>
#include <stdint.h>

typedef __attribute__((ext_vector_type(8))) int v8i;

#define TPB 256
#define BM 128
#define BN 256
#define BK 64
#define LDSK 80            // padded row stride in LDS (bank-conflict mitigation)
#define XCAP 32            // max tracked outlier columns
#define THRESH 6.0f

// ---------------- CDNA5 async global->LDS copy (inline asm) -----------------
__device__ __forceinline__ void cp_async16(unsigned lds_off, const void* gptr) {
  // GV mode: vdst = LDS byte address, vaddr = 64-bit global address
  asm volatile("global_load_async_to_lds_b128 %0, %1, off"
               :: "v"(lds_off), "v"(gptr) : "memory");
}
__device__ __forceinline__ void wait_async0() {
  asm volatile("s_wait_asynccnt 0" ::: "memory");
}

// ---------------- Stage 1: per-row weight quantization ----------------------
__global__ __launch_bounds__(TPB) void k_quantw(const float* __restrict__ W,
                                                signed char* __restrict__ CB,
                                                float* __restrict__ SCB,
                                                int O, int D) {
  int row = blockIdx.x;
  if (row >= O) return;
  const float* w = W + (size_t)row * D;
  float m = 0.0f;
  for (int d = threadIdx.x; d < D; d += TPB) m = fmaxf(m, fabsf(w[d]));
  __shared__ float red[TPB];
  red[threadIdx.x] = m; __syncthreads();
  for (int s = TPB / 2; s > 0; s >>= 1) {
    if (threadIdx.x < s) red[threadIdx.x] = fmaxf(red[threadIdx.x], red[threadIdx.x + s]);
    __syncthreads();
  }
  float scb = red[0];
  if (threadIdx.x == 0) SCB[row] = scb;
  float inv = 127.0f / fmaxf(scb, 1e-8f);
  signed char* cb = CB + (size_t)row * D;
  for (int d = threadIdx.x; d < D; d += TPB) {
    float q = rintf(w[d] * inv);             // round-half-even, like jnp.round
    q = fminf(fmaxf(q, -127.0f), 127.0f);
    cb[d] = (signed char)q;
  }
}

// ---------------- Stage 2: outlier-column detection -------------------------
__global__ void k_zero_aux(int* __restrict__ mask, int* __restrict__ count, int D) {
  int i = blockIdx.x * blockDim.x + threadIdx.x;
  if (i < D) mask[i] = 0;
  if (i == 0) *count = 0;
}

__global__ void k_detect(const float* __restrict__ x, int* __restrict__ mask,
                         int N, int D) {
  size_t total = (size_t)N * D;
  size_t stride = (size_t)gridDim.x * blockDim.x;
  for (size_t idx = (size_t)blockIdx.x * blockDim.x + threadIdx.x; idx < total; idx += stride) {
    float v = x[idx];
    if (fabsf(v) > THRESH) {
      size_t col = idx - (idx / D) * D;      // div only on the rare path
      mask[col] = 1;
    }
  }
}

// deterministic ascending compaction (tiny: D iterations by one thread)
__global__ void k_compact(const int* __restrict__ mask, int* __restrict__ cols,
                          int* __restrict__ count, int D) {
  if (blockIdx.x == 0 && threadIdx.x == 0) {
    int c = 0;
    for (int d = 0; d < D; ++d)
      if (mask[d] && c < XCAP) cols[c++] = d;
    *count = c;
  }
}

// ---------------- Stage 3: per-token activation quantization ----------------
__global__ __launch_bounds__(TPB) void k_quantx(const float* __restrict__ x,
                                                const int* __restrict__ mask,
                                                const int* __restrict__ cols,
                                                const int* __restrict__ count,
                                                signed char* __restrict__ Cx,
                                                float* __restrict__ rowmax,
                                                float* __restrict__ Xc,
                                                int N, int D) {
  int row = blockIdx.x;
  if (row >= N) return;
  const float* xr = x + (size_t)row * D;
  float m = 0.0f;
  for (int d = threadIdx.x; d < D; d += TPB) {
    if (!mask[d]) m = fmaxf(m, fabsf(xr[d]));
  }
  __shared__ float red[TPB];
  red[threadIdx.x] = m; __syncthreads();
  for (int s = TPB / 2; s > 0; s >>= 1) {
    if (threadIdx.x < s) red[threadIdx.x] = fmaxf(red[threadIdx.x], red[threadIdx.x + s]);
    __syncthreads();
  }
  float absmax = fmaxf(red[0], 1e-8f);
  if (threadIdx.x == 0) rowmax[row] = absmax;
  float inv = 127.0f / absmax;
  signed char* cx = Cx + (size_t)row * D;
  for (int d = threadIdx.x; d < D; d += TPB) {
    float v = mask[d] ? 0.0f : xr[d];
    float q = rintf(v * inv);
    q = fminf(fmaxf(q, -127.0f), 127.0f);
    cx[d] = (signed char)q;
  }
  int K = *count;
  for (int t = threadIdx.x; t < XCAP; t += TPB)
    Xc[(size_t)row * XCAP + t] = (t < K) ? xr[cols[t]] : 0.0f;
}

// ---------------- Stage 4: int8 WMMA GEMM + fused dequant/outlier/bias ------
// ISA 8-bit A-fragment layout (16x64): lanes 0-15 hold rows, 8-byte K-chunks at
// offsets {0,16,32,48}; lanes 16-31 same rows with +8 K-offset. B uses the
// mirrored layout with lanes = output columns (both operands are K-major).
__device__ __forceinline__ v8i frag_ld(const signed char* __restrict__ t,
                                       int row0, int lane) {
  const int hi = lane >> 4;
  const signed char* p = t + (size_t)(row0 + (lane & 15)) * LDSK + hi * 8;
  v8i f;
#pragma unroll
  for (int c = 0; c < 4; ++c) {
    int2 d = *(const int2*)(p + c * 16);
    f[2 * c]     = d.x;
    f[2 * c + 1] = d.y;
  }
  return f;
}

// 64x64 wave tile: 4x4 WMMA tiles -> 16 WMMAs per 8 fragment loads
__device__ __forceinline__ void do_tiles(const signed char* __restrict__ At,
                                         const signed char* __restrict__ Bt,
                                         int wm, int wn, int lane,
                                         v8i acc[4][4]) {
  v8i af[4], bf[4];
#pragma unroll
  for (int i = 0; i < 4; ++i) af[i] = frag_ld(At, wm * 64 + i * 16, lane);
#pragma unroll
  for (int j = 0; j < 4; ++j) bf[j] = frag_ld(Bt, wn * 64 + j * 16, lane);
#pragma unroll
  for (int i = 0; i < 4; ++i)
#pragma unroll
    for (int j = 0; j < 4; ++j)
      acc[i][j] = __builtin_amdgcn_wmma_i32_16x16x64_iu8(
          /*sgn_a=*/true, af[i], /*sgn_b=*/true, bf[j], acc[i][j],
          /*reuse_a=*/false, /*reuse_b=*/false);
}

// stage one K-step (A: 128x64, B: 256x64 int8) into the given LDS buffers
__device__ __forceinline__ void issue_tile(unsigned aB, unsigned bB,
                                           const signed char* __restrict__ gA,
                                           const signed char* __restrict__ gB,
                                           int D, int tid, int k0) {
  const signed char* ga = gA + k0;
  const signed char* gb = gB + k0;
#pragma unroll
  for (int c = 0; c < 2; ++c) {            // A: 512 x 16B chunks
    int id = tid + c * TPB;
    int row = id >> 2;
    int off = (id & 3) * 16;
    cp_async16(aB + row * LDSK + off, ga + (size_t)row * D + off);
  }
#pragma unroll
  for (int c = 0; c < 4; ++c) {            // B: 1024 x 16B chunks
    int id = tid + c * TPB;
    int row = id >> 2;
    int off = (id & 3) * 16;
    cp_async16(bB + row * LDSK + off, gb + (size_t)row * D + off);
  }
}

__global__ __launch_bounds__(TPB) void k_gemm_i8(
    const signed char* __restrict__ Cx, const signed char* __restrict__ CB,
    const float* __restrict__ rowmax, const float* __restrict__ SCB,
    const float* __restrict__ bias, const float* __restrict__ Xc,
    const int* __restrict__ cols, const int* __restrict__ countp,
    float* __restrict__ out, int N, int D, int O) {
  __shared__ __align__(16) signed char As[2][BM * LDSK];
  __shared__ __align__(16) signed char Bs[2][BN * LDSK];

  const int tid  = threadIdx.x;
  const int lane = tid & 31;
  const int wave = tid >> 5;          // 0..7
  const int wm   = wave & 1;          // 2 waves along M  -> 64 rows each
  const int wn   = wave >> 1;         // 4 waves along N  -> 64 cols each
  const int m0   = blockIdx.y * BM;
  const int n0   = blockIdx.x * BN;

  const signed char* gA = Cx + (size_t)m0 * D;
  const signed char* gB = CB + (size_t)n0 * D;

  const unsigned a0 = (unsigned)(uintptr_t)&As[0][0];
  const unsigned a1 = (unsigned)(uintptr_t)&As[1][0];
  const unsigned b0 = (unsigned)(uintptr_t)&Bs[0][0];
  const unsigned b1 = (unsigned)(uintptr_t)&Bs[1][0];

  // prologue: stage K-step 0 into buffer 0
  issue_tile(a0, b0, gA, gB, D, tid, 0);
  wait_async0();
  __syncthreads();

  v8i acc[4][4];
#pragma unroll
  for (int i = 0; i < 4; ++i)
#pragma unroll
    for (int j = 0; j < 4; ++j)
#pragma unroll
      for (int r = 0; r < 8; ++r) acc[i][j][r] = 0;

  const int ksteps = D / BK;          // 64 for D=4096 (even)
  for (int ks = 0; ks < ksteps; ks += 2) {
    // ---- phase 0: compute buffer 0 (tile ks), prefetch tile ks+1 -> buf 1
    if (ks + 1 < ksteps) issue_tile(a1, b1, gA, gB, D, tid, (ks + 1) * BK);
    do_tiles(&As[0][0], &Bs[0][0], wm, wn, lane, acc);
    wait_async0();
    __syncthreads();

    // ---- phase 1: compute buffer 1 (tile ks+1), prefetch tile ks+2 -> buf 0
    if (ks + 1 < ksteps) {
      if (ks + 2 < ksteps) issue_tile(a0, b0, gA, gB, D, tid, (ks + 2) * BK);
      do_tiles(&As[1][0], &Bs[1][0], wm, wn, lane, acc);
      wait_async0();
      __syncthreads();
    }
  }

  // epilogue: dequant + outlier fp correction + bias
  const int K = *countp;
#pragma unroll
  for (int i = 0; i < 4; ++i) {
    int rowbase = m0 + wm * 64 + i * 16 + (lane >> 4) * 8;
#pragma unroll
    for (int j = 0; j < 4; ++j) {
      int col   = n0 + wn * 64 + j * 16 + (lane & 15);
      float sb  = SCB[col] * (1.0f / 127.0f);
      float bcl = bias[col];
      float res[8];
#pragma unroll
      for (int r = 0; r < 8; ++r) {
        float sa = rowmax[rowbase + r] * (1.0f / 127.0f);
        res[r] = (float)acc[i][j][r] * sa * sb + bcl;
      }
      for (int t = 0; t < K; ++t) {
        float wt = (float)CB[(size_t)col * D + cols[t]] * sb;   // dequantized W column
#pragma unroll
        for (int r = 0; r < 8; ++r)
          res[r] += Xc[(size_t)(rowbase + r) * XCAP + t] * wt;
      }
#pragma unroll
      for (int r = 0; r < 8; ++r)
        out[(size_t)(rowbase + r) * O + col] = res[r];
    }
  }
}

// ---------------------------------------------------------------------------
extern "C" void kernel_launch(void* const* d_in, const int* in_sizes, int n_in,
                              void* d_out, int out_size, void* d_ws, size_t ws_size,
                              hipStream_t stream) {
  const float* x    = (const float*)d_in[0];
  const float* W    = (const float*)d_in[1];
  const float* bias = (const float*)d_in[2];
  float* out        = (float*)d_out;

  const int O = in_sizes[2];               // 4096
  const int D = in_sizes[1] / O;           // 4096
  const int N = in_sizes[0] / D;           // B*S = 4096

  // workspace carve-out (256B aligned regions)
  char* p = (char*)d_ws;
  auto carve = [&](size_t bytes) -> void* {
    void* r = (void*)p;
    p += (bytes + 255) & ~(size_t)255;
    return r;
  };
  signed char* CB   = (signed char*)carve((size_t)O * D);
  signed char* Cx   = (signed char*)carve((size_t)N * D);
  float* SCB        = (float*)carve(sizeof(float) * (size_t)O);
  float* rowmax     = (float*)carve(sizeof(float) * (size_t)N);
  int*   mask       = (int*)carve(sizeof(int) * (size_t)D);
  int*   cols       = (int*)carve(sizeof(int) * XCAP);
  int*   count      = (int*)carve(sizeof(int));
  float* Xc         = (float*)carve(sizeof(float) * (size_t)N * XCAP);

  k_quantw<<<O, TPB, 0, stream>>>(W, CB, SCB, O, D);
  k_zero_aux<<<(D + TPB - 1) / TPB, TPB, 0, stream>>>(mask, count, D);
  k_detect<<<2048, TPB, 0, stream>>>(x, mask, N, D);
  k_compact<<<1, 32, 0, stream>>>(mask, cols, count, D);
  k_quantx<<<N, TPB, 0, stream>>>(x, mask, cols, count, Cx, rowmax, Xc, N, D);

  dim3 grid(O / BN, N / BM);
  k_gemm_i8<<<grid, TPB, 0, stream>>>(Cx, CB, rowmax, SCB, bias, Xc, cols, count,
                                      out, N, D, O);
}
